// KronQRLinearLayer3_cayley_88596585382841
// MI455X (gfx1250) — matrix operand order repaired and checked by reading, not compiled, not run on hardware
//
#include <hip/hip_runtime.h>
#include <stdint.h>

typedef __attribute__((ext_vector_type(16))) __bf16        v16bf;
typedef __attribute__((ext_vector_type(8)))  float         v8f;
typedef __attribute__((ext_vector_type(8)))  unsigned int  v8u;
typedef __attribute__((ext_vector_type(4)))  unsigned int  v4u;

// ---------- bf16 split helpers (round-to-nearest-even) ----------
__device__ __forceinline__ unsigned int f32_to_bf16_rne(float f) {
  unsigned int u = __builtin_bit_cast(unsigned int, f);
  return (u + 0x7FFFu + ((u >> 16) & 1u)) >> 16;
}
__device__ __forceinline__ float bf16_to_f32(unsigned int b) {
  return __builtin_bit_cast(float, b << 16);
}
__device__ __forceinline__ void split1(float f, unsigned short& h, unsigned short& l) {
  unsigned int hb = f32_to_bf16_rne(f);
  unsigned int lb = f32_to_bf16_rne(f - bf16_to_f32(hb));
  h = (unsigned short)hb;
  l = (unsigned short)lb;
}

// Split 16 consecutive fp32 operand values into hi/lo bf16 fragments.
__device__ __forceinline__ void split16(const float* v, v16bf& hi, v16bf& lo) {
  v8u hb, lb;
#pragma unroll
  for (int r = 0; r < 8; ++r) {
    float f0 = v[2 * r], f1 = v[2 * r + 1];
    unsigned int h0 = f32_to_bf16_rne(f0);
    unsigned int h1 = f32_to_bf16_rne(f1);
    float r0 = f0 - bf16_to_f32(h0);
    float r1 = f1 - bf16_to_f32(h1);
    unsigned int l0 = f32_to_bf16_rne(r0);
    unsigned int l1 = f32_to_bf16_rne(r1);
    hb[r] = (h0 & 0xFFFFu) | (h1 << 16);
    lb[r] = (l0 & 0xFFFFu) | (l1 << 16);
  }
  hi = __builtin_bit_cast(v16bf, hb);
  lo = __builtin_bit_cast(v16bf, lb);
}

// ---------- Kernel 1: Cayley transforms (4x4, 8x8, 40x40) ----------
#define Q1_OFF 0
#define Q2_OFF 16
#define Q3_OFF 80

__device__ void do_cayley(const float* A, int n, float* qout,
                          float* aug, float* nm, float* fac) {
  const int t = threadIdx.x, nt = blockDim.x;
  const int w = 2 * n;
  for (int idx = t; idx < n * n; idx += nt) {
    int i = idx / n, j = idx - i * n;
    float s = 0.5f * (A[i * n + j] - A[j * n + i]);
    float d = (i == j) ? 1.0f : 0.0f;
    aug[i * w + j]     = d + s;
    aug[i * w + n + j] = d;
    nm[idx]            = d - s;
  }
  __syncthreads();
  // Gauss-Jordan (I+skew is normal, eigenvalues 1±i*lam -> no pivoting needed)
  for (int p = 0; p < n; ++p) {
    float pinv = 1.0f / aug[p * w + p];
    __syncthreads();
    for (int j = t; j < w; j += nt) aug[p * w + j] *= pinv;
    __syncthreads();
    for (int r = t; r < n; r += nt) fac[r] = aug[r * w + p];
    __syncthreads();
    for (int idx = t; idx < n * w; idx += nt) {
      int r = idx / w, j = idx - r * w;
      if (r != p) aug[idx] -= fac[r] * aug[p * w + j];
    }
    __syncthreads();
  }
  for (int idx = t; idx < n * n; idx += nt) {
    int i = idx / n, j = idx - i * n;
    float s = 0.0f;
    for (int kk = 0; kk < n; ++kk) s += nm[i * n + kk] * aug[kk * w + n + j];
    qout[idx] = s;
  }
  __syncthreads();
}

__global__ __launch_bounds__(64) void cayley_kernel(const float* k1, const float* k2,
                                                    const float* k3, float* q) {
  __shared__ float aug[40 * 80];
  __shared__ float nm[40 * 40];
  __shared__ float fac[40];
  do_cayley(k1, 4,  q + Q1_OFF, aug, nm, fac);
  do_cayley(k2, 8,  q + Q2_OFF, aug, nm, fac);
  do_cayley(k3, 40, q + Q3_OFF, aug, nm, fac);
}

// ---------- Kernel 2: R = kron(kron(q1,q2),q3), emitted pre-split bf16 hi/lo ----------
__global__ __launch_bounds__(256) void kron_split_kernel(const float* __restrict__ q,
                                                         unsigned short* __restrict__ rhi,
                                                         unsigned short* __restrict__ rlo) {
  int idx = blockIdx.x * 256 + threadIdx.x;
  if (idx >= 1280 * 1280) return;
  int i = idx / 1280, j = idx - i * 1280;
  int i1 = i / 320, i2 = (i / 40) & 7, i3 = i % 40;
  int j1 = j / 320, j2 = (j / 40) & 7, j3 = j % 40;
  float r = q[Q1_OFF + i1 * 4 + j1] * q[Q2_OFF + i2 * 8 + j2] * q[Q3_OFF + i3 * 40 + j3];
  split1(r, rhi[idx], rlo[idx]);
}

// ---------- Kernel 3: fp32 -> bf16 hi/lo split, 4 elements/thread ----------
__global__ __launch_bounds__(256) void split4_kernel(const float* __restrict__ src,
                                                     unsigned short* __restrict__ hi,
                                                     unsigned short* __restrict__ lo,
                                                     long n4) {
  long idx = (long)blockIdx.x * 256 + threadIdx.x;
  if (idx >= n4) return;
  float4 f = ((const float4*)src)[idx];
  ushort4 h, l;
  split1(f.x, h.x, l.x);
  split1(f.y, h.y, l.y);
  split1(f.z, h.z, l.z);
  split1(f.w, h.w, l.w);
  ((ushort4*)hi)[idx] = h;
  ((ushort4*)lo)[idx] = l;
}

// ================= WMMA GEMM: C[m,n] = sum_k A[m,k]*B2[n,k] =================
// Block: 256 threads = 8 wave32, block tile 128(M) x 128(N).
// Wave: 32(M) x 64(N) = 2x4 tiles of v_wmma_f32_16x16x32_bf16, bf16x3 emulated fp32.

// Variant A: both operands pre-split bf16 (near-zero inner-loop VALU).
__global__ __launch_bounds__(256) void gemm_bb_wmma(
    const unsigned short* __restrict__ Ahi, const unsigned short* __restrict__ Alo,
    const unsigned short* __restrict__ Bhi, const unsigned short* __restrict__ Blo,
    float* __restrict__ C, int M, int N, int K) {
  const int lane  = threadIdx.x & 31;
  const int wid   = threadIdx.x >> 5;
  const int wm    = wid & 3;
  const int wn    = wid >> 2;
  const int bm    = blockIdx.y * 128 + wm * 32;
  const int bn    = blockIdx.x * 128 + wn * 64;

  const int arow  = lane & 15;
  const int ahalf = lane >> 4;
  const int akc   = ahalf * 16;   // A: 16 contiguous bf16 at akc
  const int bkc   = ahalf * 16;   // B: 16 contiguous bf16 at bkc

  v8f acc[2][4] = {};

  for (int k = 0; k < K; k += 32) {
    v16bf ah[2], al[2];
#pragma unroll
    for (int tm = 0; tm < 2; ++tm) {
      size_t off = (size_t)(bm + tm * 16 + arow) * K + k + akc;
      v4u h0 = *(const v4u*)(Ahi + off);
      v4u h1 = *(const v4u*)(Ahi + off + 8);
      v4u l0 = *(const v4u*)(Alo + off);
      v4u l1 = *(const v4u*)(Alo + off + 8);
      ah[tm] = __builtin_bit_cast(v16bf,
                 __builtin_shufflevector(h0, h1, 0, 1, 2, 3, 4, 5, 6, 7));
      al[tm] = __builtin_bit_cast(v16bf,
                 __builtin_shufflevector(l0, l1, 0, 1, 2, 3, 4, 5, 6, 7));
    }
    v16bf bh[4], bl[4];
#pragma unroll
    for (int tn = 0; tn < 4; ++tn) {
      size_t off = (size_t)(bn + tn * 16 + arow) * K + k + bkc;
      bh[tn] = *(const v16bf*)(Bhi + off);
      bl[tn] = *(const v16bf*)(Blo + off);
    }
#pragma unroll
    for (int tm = 0; tm < 2; ++tm)
#pragma unroll
      for (int tn = 0; tn < 4; ++tn) {
        acc[tm][tn] = __builtin_amdgcn_wmma_f32_16x16x32_bf16(
            false, ah[tm], false, bh[tn], (short)0, acc[tm][tn], false, false);
        acc[tm][tn] = __builtin_amdgcn_wmma_f32_16x16x32_bf16(
            false, ah[tm], false, bl[tn], (short)0, acc[tm][tn], false, false);
        acc[tm][tn] = __builtin_amdgcn_wmma_f32_16x16x32_bf16(
            false, al[tm], false, bh[tn], (short)0, acc[tm][tn], false, false);
      }
  }

  const int crow0 = ahalf * 8;
  const int ccol  = lane & 15;
#pragma unroll
  for (int tm = 0; tm < 2; ++tm)
#pragma unroll
    for (int tn = 0; tn < 4; ++tn) {
      size_t base = (size_t)(bm + tm * 16 + crow0) * N + (size_t)(bn + tn * 16 + ccol);
#pragma unroll
      for (int r = 0; r < 8; ++r)
        C[base + (size_t)r * N] = acc[tm][tn][r];
    }
}

// Variant B (fallback for small ws): A fp32 with on-the-fly split.
__global__ __launch_bounds__(256) void gemm_nt_wmma(
    const float* __restrict__ A, const unsigned short* __restrict__ Bhi,
    const unsigned short* __restrict__ Blo, float* __restrict__ C,
    int M, int N, int K) {
  const int lane  = threadIdx.x & 31;
  const int wid   = threadIdx.x >> 5;
  const int wm    = wid & 3;
  const int wn    = wid >> 2;
  const int bm    = blockIdx.y * 128 + wm * 32;
  const int bn    = blockIdx.x * 128 + wn * 64;

  const int arow  = lane & 15;
  const int ahalf = lane >> 4;
  const int akc   = ahalf * 16;
  const int bkc   = ahalf * 16;

  v8f acc[2][4] = {};

  for (int k = 0; k < K; k += 32) {
    v16bf ah[2], al[2];
#pragma unroll
    for (int tm = 0; tm < 2; ++tm) {
      const float* p0 = A + (size_t)(bm + tm * 16 + arow) * K + k + akc;
      float v[16];
      const float4 c0 = *(const float4*)(p0);
      const float4 c1 = *(const float4*)(p0 + 4);
      const float4 c2 = *(const float4*)(p0 + 8);
      const float4 c3 = *(const float4*)(p0 + 12);
      v[0]=c0.x;  v[1]=c0.y;  v[2]=c0.z;  v[3]=c0.w;
      v[4]=c1.x;  v[5]=c1.y;  v[6]=c1.z;  v[7]=c1.w;
      v[8]=c2.x;  v[9]=c2.y;  v[10]=c2.z; v[11]=c2.w;
      v[12]=c3.x; v[13]=c3.y; v[14]=c3.z; v[15]=c3.w;
      split16(v, ah[tm], al[tm]);
    }
    v16bf bh[4], bl[4];
#pragma unroll
    for (int tn = 0; tn < 4; ++tn) {
      size_t off = (size_t)(bn + tn * 16 + arow) * K + k + bkc;
      bh[tn] = *(const v16bf*)(Bhi + off);
      bl[tn] = *(const v16bf*)(Blo + off);
    }
#pragma unroll
    for (int tm = 0; tm < 2; ++tm)
#pragma unroll
      for (int tn = 0; tn < 4; ++tn) {
        acc[tm][tn] = __builtin_amdgcn_wmma_f32_16x16x32_bf16(
            false, ah[tm], false, bh[tn], (short)0, acc[tm][tn], false, false);
        acc[tm][tn] = __builtin_amdgcn_wmma_f32_16x16x32_bf16(
            false, ah[tm], false, bl[tn], (short)0, acc[tm][tn], false, false);
        acc[tm][tn] = __builtin_amdgcn_wmma_f32_16x16x32_bf16(
            false, al[tm], false, bh[tn], (short)0, acc[tm][tn], false, false);
      }
  }

  const int crow0 = ahalf * 8;
  const int ccol  = lane & 15;
#pragma unroll
  for (int tm = 0; tm < 2; ++tm)
#pragma unroll
    for (int tn = 0; tn < 4; ++tn) {
      size_t base = (size_t)(bm + tm * 16 + crow0) * N + (size_t)(bn + tn * 16 + ccol);
#pragma unroll
      for (int r = 0; r < 8; ++r)
        C[base + (size_t)r * N] = acc[tm][tn][r];
    }
}

// ---------- Host launch ----------
extern "C" void kernel_launch(void* const* d_in, const int* in_sizes, int n_in,
                              void* d_out, int out_size, void* d_ws, size_t ws_size,
                              hipStream_t stream) {
  const float* x  = (const float*)d_in[0];
  const float* k1 = (const float*)d_in[1];
  const float* k2 = (const float*)d_in[2];
  const float* k3 = (const float*)d_in[3];
  const float* W  = (const float*)d_in[4];
  float* out = (float*)d_out;

  const size_t NE = 1280ull * 1280ull;     // filt/R/W elements
  const size_t XE = 32768ull * 1280ull;    // x elements

  // ws layout
  float* qbuf = (float*)d_ws;                       // 4096 floats
  float* filt = qbuf + 4096;                        // NE fp32
  unsigned short* Rhi = (unsigned short*)(filt + NE);
  unsigned short* Rlo = Rhi + NE;
  unsigned short* Fhi = Rlo + NE;
  unsigned short* Flo = Fhi + NE;
  unsigned short* Whi = Flo + NE;                   // full path only
  unsigned short* Wlo = Whi + NE;
  unsigned short* Xhi = Wlo + NE;
  unsigned short* Xlo = Xhi + XE;
  const size_t need_full = (size_t)((char*)(Xlo + XE) - (char*)d_ws);   // ~185 MB
  const bool full = (ws_size >= need_full);

  // 1) tiny Cayley transforms
  cayley_kernel<<<1, 64, 0, stream>>>(k1, k2, k3, qbuf);
  // 2) R = q1 (x) q2 (x) q3, pre-split to bf16 hi/lo
  kron_split_kernel<<<(int)((NE + 255) / 256), 256, 0, stream>>>(qbuf, Rhi, Rlo);

  if (full) {
    // pre-split W and x once: kills the 10x-redundant in-register conversions
    split4_kernel<<<(int)((NE / 4 + 255) / 256), 256, 0, stream>>>(W, Whi, Wlo, (long)(NE / 4));
    split4_kernel<<<(int)((XE / 4 + 255) / 256), 256, 0, stream>>>(x, Xhi, Xlo, (long)(XE / 4));
    // 3) filt = W @ R^T
    gemm_bb_wmma<<<dim3(10, 10), 256, 0, stream>>>(Whi, Wlo, Rhi, Rlo, filt, 1280, 1280, 1280);
    // 4) split filt for reuse as WMMA B operand
    split4_kernel<<<(int)((NE / 4 + 255) / 256), 256, 0, stream>>>(filt, Fhi, Flo, (long)(NE / 4));
    // 5) out = x @ filt^T  (107 GFLOP bulk, bf16x3 emulated fp32)
    gemm_bb_wmma<<<dim3(10, 256), 256, 0, stream>>>(Xhi, Xlo, Fhi, Flo, out, 32768, 1280, 1280);
  } else {
    gemm_nt_wmma<<<dim3(10, 10), 256, 0, stream>>>(W, Rhi, Rlo, filt, 1280, 1280, 1280);
    split4_kernel<<<(int)((NE / 4 + 255) / 256), 256, 0, stream>>>(filt, Fhi, Flo, (long)(NE / 4));
    gemm_nt_wmma<<<dim3(10, 256), 256, 0, stream>>>(x, Fhi, Flo, out, 32768, 1280, 1280);
  }
}
